// SelectiveSSM_78958678769751
// MI455X (gfx1250) — compile-verified, matrix operand order checked by reading, not run on hardware
//
#include <hip/hip_runtime.h>
#include <hip/hip_bf16.h>

// ---------------------------------------------------------------------------
// Selective SSM forward for MI455X (gfx1250, wave32, WMMA + TDM + async-LDS).
// GEMMs on v_wmma_f32_16x16x32_bf16. Operands pre-converted to bf16 once, so
// GEMM tile staging is pure data movement: B tiles via tensor_load_to_lds
// (TDM, padded-LDS descriptor), A tiles via global_load_async_to_lds_b128.
// GEMM2 fuses y = (ys + xc*D)*silu(z) into its A-tile loader (VALU staging).
// ---------------------------------------------------------------------------

typedef __attribute__((ext_vector_type(16))) __bf16 v16bf;
typedef __attribute__((ext_vector_type(8)))  __bf16 v8bf;
typedef __attribute__((ext_vector_type(4)))  __bf16 v4bf;
typedef __attribute__((ext_vector_type(8)))  float  v8f;
typedef __attribute__((ext_vector_type(4)))  unsigned int u32x4;
typedef __attribute__((ext_vector_type(8)))  int i32x8;
typedef __attribute__((ext_vector_type(4)))  int i32x4;

#define D_MODEL  1024
#define D_INNER  2048
#define NSTATE   16
#define BATCH    4
#define SEQ      4096
#define M_TOTAL  (BATCH * SEQ)     // 16384

// GEMM tiling: 128x128 block tile, BK=32, 8 waves (2 along M x 4 along N),
// each wave owns a 64x32 tile = 4x2 WMMA fragments.
#define BM  128
#define BN  128
#define BK  32
#define LDT 40                     // padded LDS row stride (bf16 elements)

__device__ __forceinline__ float sigmoidf_(float v) {
    return 1.0f / (1.0f + __expf(-v));
}
__device__ __forceinline__ float siluf_(float v) { return v * sigmoidf_(v); }

__device__ __forceinline__ v16bf mk16(v8bf lo, v8bf hi) {
    return __builtin_shufflevector(lo, hi, 0, 1, 2, 3, 4, 5, 6, 7,
                                           8, 9, 10, 11, 12, 13, 14, 15);
}

__device__ __forceinline__ float red16(float v) {
    v += __shfl_xor(v, 1, 32);
    v += __shfl_xor(v, 2, 32);
    v += __shfl_xor(v, 4, 32);
    v += __shfl_xor(v, 8, 32);
    return v;
}

// Generic->LDS byte offset: the LDS offset is the low 32 bits of the flat address.
__device__ __forceinline__ unsigned int lds_off(const void* p) {
    return (unsigned int)(unsigned long long)p;
}

// ---------------------------------------------------------------------------
// TDM: load a [rows=128, cols=BK(bf16)] tile of a row-major bf16 matrix with
// row stride K elements into LDS with the LDT=40 padded layout.
//   data_size=2B, tile_dim0=32, tile_dim1=128, tensor_dim0_stride=K
//   pad_enable, pad_interval=16 DWORDs (one 64B row), pad_amount=4 DWORDs
// ---------------------------------------------------------------------------
__device__ __forceinline__ void tdm_load_tile(unsigned int ldsoff,
                                              const __bf16* gsrc, int K) {
    const unsigned long long ga = (unsigned long long)gsrc;
    u32x4 g0;
    g0[0] = 1u;                                      // count=1
    g0[1] = ldsoff;                                  // lds_addr (bytes)
    g0[2] = (unsigned int)ga;                        // global_addr[31:0]
    g0[3] = (unsigned int)(ga >> 32) | (2u << 30);   // addr[56:32] | type=2
    i32x8 g1;
    g1[0] = (int)((1u << 16) | (1u << 20) | (3u << 22) | (3u << 25));
    g1[1] = (int)(((unsigned)K & 0xffffu) << 16);    // tensor_dim0 lo16
    g1[2] = (int)((unsigned)K >> 16);                // tensor_dim0 hi16 | td1 lo
    g1[3] = (int)(1u | (32u << 16));                 // td1 hi (td1=65536) | tile_dim0=32
    g1[4] = 128;                                     // tile_dim1=128, tile_dim2=0
    g1[5] = K;                                       // tensor_dim0_stride lo32
    g1[6] = 0;
    g1[7] = 0;
    i32x4 gz4 = {0, 0, 0, 0};
    i32x8 gz8 = {0, 0, 0, 0, 0, 0, 0, 0};
    __builtin_amdgcn_tensor_load_to_lds(g0, g1, gz4, gz4, gz8, 0);
}

// ---------------------------------------------------------------------------
// f32 -> bf16 bulk conversion (8 elements/thread)
// ---------------------------------------------------------------------------
__global__ __launch_bounds__(256) void cvt_bf16_kernel(
    const float* __restrict__ src, __bf16* __restrict__ dst)
{
    const size_t i = ((size_t)blockIdx.x * 256 + threadIdx.x) * 8;
    float4 a = *(const float4*)(src + i);
    float4 b = *(const float4*)(src + i + 4);
    v8bf o;
    o[0] = (__bf16)a.x; o[1] = (__bf16)a.y; o[2] = (__bf16)a.z; o[3] = (__bf16)a.w;
    o[4] = (__bf16)b.x; o[5] = (__bf16)b.y; o[6] = (__bf16)b.z; o[7] = (__bf16)b.w;
    *(v8bf*)(dst + i) = o;
}

// ---------------------------------------------------------------------------
// Tiled WMMA GEMM:  D[M][Nout] = A[M][K] * B[Nout][K]^T   (bf16 operands)
// MODE 0: A = bf16 matrix, staged via async-to-LDS; output bf16.
// MODE 1: A[m][k] = (ys[m] + xc[m][k]*D[k]) * silu(z[m][k]) on the fly
//         (VALU staging); output f32.
// B tiles staged via TDM in both modes.
// ---------------------------------------------------------------------------
template <int MODE>
__global__ __launch_bounds__(256) void gemm_wmma(
    const __bf16* __restrict__ Abf,     // MODE0: bf16 A [M][K]
    const __bf16* __restrict__ Bbf,     // bf16 W row-major [Nout][K]
    __bf16* __restrict__ out_bf,        // MODE0 output
    float* __restrict__ out_f,          // MODE1 output
    const __bf16* __restrict__ xca,     // MODE1: xc [M][2048]
    const __bf16* __restrict__ xz,      // MODE1: z at xz[m][2048+k]
    const float* __restrict__ ysv,      // MODE1: scan output [M]
    const float* __restrict__ Dv,       // MODE1: D [2048]
    int K, int Nout)
{
    __shared__ __bf16 sA[BM * LDT];
    __shared__ __bf16 sB[BN * LDT];

    const int tid  = threadIdx.x;
    const int m0   = blockIdx.y * BM;
    const int n0   = blockIdx.x * BN;
    const int wave = tid >> 5;
    const int lane = tid & 31;
    const int wm   = wave >> 2;        // 0..1
    const int wn   = wave & 3;         // 0..3
    const int lr   = lane & 15;
    const int lh   = lane >> 4;

    // Cooperative staging: 2 threads per row, 16 bf16 (32B) each.
    const int r    = tid >> 1;         // 0..127
    const int cseg = (tid & 1) * 16;   // 0 or 16 (elements)

    v8f acc[4][2] = {};

    for (int k0 = 0; k0 < K; k0 += BK) {
        __syncthreads();   // all waves done reading LDS from previous tile

        // ---- B tile via Tensor Data Mover (wave 0 issues, TENSORcnt) ----
        if (wave == 0) {
            tdm_load_tile(lds_off(&sB[0]), Bbf + (size_t)n0 * K + k0, K);
        }

        // ---- A tile ----
        if (MODE == 0) {
            // pure copy: async global->LDS, 2 x b128 per thread
            const unsigned int la = lds_off(&sA[r * LDT + cseg]);
            const __bf16* gp = Abf + (size_t)(m0 + r) * K + k0 + cseg;
            asm volatile("global_load_async_to_lds_b128 %0, %1, off"
                         :: "v"(la), "v"((unsigned long long)gp) : "memory");
            asm volatile("global_load_async_to_lds_b128 %0, %1, off offset:16"
                         :: "v"(la), "v"((unsigned long long)gp) : "memory");
            asm volatile("s_wait_asynccnt 0x0" ::: "memory");
        } else {
            // fused y = (ys + xc*D) * silu(z), converted to bf16 into LDS
            const int   gm  = m0 + r;
            const float ysr = ysv[gm];
            const size_t bx = (size_t)gm * D_INNER + k0 + cseg;
            const size_t bz = (size_t)gm * (2 * D_INNER) + D_INNER + k0 + cseg;
#pragma unroll
            for (int q = 0; q < 4; ++q) {
                v4bf xc4 = *(const v4bf*)(xca + bx + q * 4);
                v4bf z4  = *(const v4bf*)(xz  + bz + q * 4);
                float4 d4 = *(const float4*)(Dv + k0 + cseg + q * 4);
                v4bf o;
                float zz;
                zz = (float)z4[0];
                o[0] = (__bf16)((ysr + (float)xc4[0] * d4.x) * (zz * sigmoidf_(zz)));
                zz = (float)z4[1];
                o[1] = (__bf16)((ysr + (float)xc4[1] * d4.y) * (zz * sigmoidf_(zz)));
                zz = (float)z4[2];
                o[2] = (__bf16)((ysr + (float)xc4[2] * d4.z) * (zz * sigmoidf_(zz)));
                zz = (float)z4[3];
                o[3] = (__bf16)((ysr + (float)xc4[3] * d4.w) * (zz * sigmoidf_(zz)));
                *(v4bf*)&sA[r * LDT + cseg + q * 4] = o;
            }
            if (k0 + BK < K) __builtin_prefetch(xca + bx + BK, 0, 1);
        }

        if (wave == 0) __builtin_amdgcn_s_wait_tensorcnt(0);
        __syncthreads();   // staged data (TDM + async + ds stores) visible

        // ---- fragments + WMMA ----
        v16bf bfr[2];
#pragma unroll
        for (int j = 0; j < 2; ++j) {
            const int col = wn * 32 + j * 16 + lr;
            v8bf lo = *(v8bf*)&sB[col * LDT + lh * 8];
            v8bf hi = *(v8bf*)&sB[col * LDT + lh * 8 + 16];
            bfr[j] = mk16(lo, hi);
        }
#pragma unroll
        for (int i = 0; i < 4; ++i) {
            const int row = wm * 64 + i * 16 + lr;
            v8bf lo = *(v8bf*)&sA[row * LDT + lh * 8];
            v8bf hi = *(v8bf*)&sA[row * LDT + lh * 8 + 16];
            v16bf af = mk16(lo, hi);
#pragma unroll
            for (int j = 0; j < 2; ++j) {
                acc[i][j] = __builtin_amdgcn_wmma_f32_16x16x32_bf16(
                    false, af, false, bfr[j], (short)0, acc[i][j], false, false);
            }
        }
    }

    // ---- epilogue ----
#pragma unroll
    for (int i = 0; i < 4; ++i) {
#pragma unroll
        for (int j = 0; j < 2; ++j) {
            const int gn = n0 + wn * 32 + j * 16 + lr;
#pragma unroll
            for (int e = 0; e < 8; ++e) {
                const int gm = m0 + wm * 64 + i * 16 + lh * 8 + e;
                const float v = acc[i][j][e];
                if (MODE == 0)
                    out_bf[(size_t)gm * Nout + gn] = (__bf16)v;
                else
                    out_f[(size_t)gm * Nout + gn] = v;
            }
        }
    }
}

// ---------------------------------------------------------------------------
// Depthwise causal conv (width 4) + bias + SiLU.  xz bf16 -> xca bf16.
// ---------------------------------------------------------------------------
__global__ __launch_bounds__(256) void conv_silu_kernel(
    const __bf16* __restrict__ xz, const float* __restrict__ cw,
    const float* __restrict__ cb, __bf16* __restrict__ xca)
{
    const int idx = blockIdx.x * 256 + threadIdx.x;    // over M * 512
    const int e0  = (idx & (D_INNER / 4 - 1)) << 2;
    const int m   = idx >> 9;
    const int l   = m & (SEQ - 1);

    float acc0 = cb[e0 + 0], acc1 = cb[e0 + 1], acc2 = cb[e0 + 2], acc3 = cb[e0 + 3];
#pragma unroll
    for (int t = 0; t < 4; ++t) {
        const int ls = l - 3 + t;
        if (ls >= 0) {
            const __bf16* p = xz + (size_t)(m - 3 + t) * (2 * D_INNER) + e0;
            acc0 += cw[(e0 + 0) * 4 + t] * (float)p[0];
            acc1 += cw[(e0 + 1) * 4 + t] * (float)p[1];
            acc2 += cw[(e0 + 2) * 4 + t] * (float)p[2];
            acc3 += cw[(e0 + 3) * 4 + t] * (float)p[3];
        }
    }
    __bf16* o = xca + (size_t)m * D_INNER + e0;
    o[0] = (__bf16)siluf_(acc0);
    o[1] = (__bf16)siluf_(acc1);
    o[2] = (__bf16)siluf_(acc2);
    o[3] = (__bf16)siluf_(acc3);
}

// ---------------------------------------------------------------------------
// x_dbl = xc @ W_x^T (33 outputs) + softplus/exp prep; one wave per row m.
// Emits dA, dBx, C in time-transposed [b][n][SEQ] layout for the scan.
// ---------------------------------------------------------------------------
__global__ __launch_bounds__(256) void xdbl_kernel(
    const __bf16* __restrict__ xca, const float* __restrict__ Wx,
    const float* __restrict__ A_log,
    float* __restrict__ dA_t, float* __restrict__ dBx_t, float* __restrict__ C_t)
{
    const int wave = threadIdx.x >> 5;
    const int lane = threadIdx.x & 31;
    const int m    = blockIdx.x * 8 + wave;

    const __bf16* xr = xca + (size_t)m * D_INNER;

    float acc[33];
#pragma unroll
    for (int f = 0; f < 33; ++f) acc[f] = 0.0f;

#pragma unroll 1
    for (int kb = 0; kb < D_INNER; kb += 32) {
        const float xv = (float)xr[kb + lane];
#pragma unroll
        for (int f = 0; f < 33; ++f)
            acc[f] += xv * Wx[f * D_INNER + kb + lane];
    }
#pragma unroll
    for (int f = 0; f < 33; ++f) {
        float v = acc[f];
        v += __shfl_xor(v, 16, 32);
        acc[f] = red16(v);
    }

    const float draw  = acc[0];
    const float delta = (draw > 20.0f) ? draw : log1pf(__expf(draw));

    if (lane < NSTATE) {
        const int n = lane;
        const float a   = -__expf(A_log[n]);
        const float dA  = __expf(delta * a);
        const float dbx = delta * acc[1 + n] * (float)xr[n];
        const int b = m >> 12;            // m / SEQ
        const int l = m & (SEQ - 1);
        const size_t o = ((size_t)(b * NSTATE + n)) * SEQ + l;
        dA_t[o]  = dA;
        dBx_t[o] = dbx;
        C_t[o]   = acc[17 + n];
    }
}

// ---------------------------------------------------------------------------
// Parallel linear recurrence: 4 batches x 16 states x 16 segments of 256.
// ---------------------------------------------------------------------------
__global__ __launch_bounds__(1024) void scan_kernel(
    const float* __restrict__ dA_t, const float* __restrict__ dBx_t,
    const float* __restrict__ C_t, float* __restrict__ ys)
{
    __shared__ float Pl[BATCH][16][NSTATE];
    __shared__ float Ql[BATCH][16][NSTATE];
    __shared__ float Hin[BATCH][16][NSTATE];

    const int tid = threadIdx.x;
    const int n   = tid & 15;
    const int s   = (tid >> 4) & 15;
    const int b   = tid >> 8;
    const int SEGL = SEQ / 16;            // 256

    const size_t row = ((size_t)(b * NSTATE + n)) * SEQ + (size_t)s * SEGL;
    const float* pa = dA_t  + row;
    const float* px = dBx_t + row;
    const float* pc = C_t   + row;

    // Pass 1: segment totals
    float h = 0.0f, P = 1.0f;
#pragma unroll 4
    for (int i = 0; i < SEGL; i += 4) {
        float4 a4 = *(const float4*)(pa + i);
        float4 x4 = *(const float4*)(px + i);
        h = a4.x * h + x4.x;
        h = a4.y * h + x4.y;
        h = a4.z * h + x4.z;
        h = a4.w * h + x4.w;
        P *= a4.x * a4.y * a4.z * a4.w;
    }
    Pl[b][s][n] = P;
    Ql[b][s][n] = h;
    __syncthreads();

    // Pass 2: sequential fixup over 16 segments (64 threads)
    if (tid < BATCH * NSTATE) {
        const int bb = tid >> 4, nn = tid & 15;
        float hh = 0.0f;
        for (int ss = 0; ss < 16; ++ss) {
            Hin[bb][ss][nn] = hh;
            hh = Pl[bb][ss][nn] * hh + Ql[bb][ss][nn];
        }
    }
    __syncthreads();

    // Pass 3: replay with correct h_in, emit y
    float hh = Hin[b][s][n];
    float* pys = ys + (size_t)b * SEQ + (size_t)s * SEGL;
#pragma unroll 2
    for (int i = 0; i < SEGL; i += 4) {
        float4 a4 = *(const float4*)(pa + i);
        float4 x4 = *(const float4*)(px + i);
        float4 c4 = *(const float4*)(pc + i);
        hh = a4.x * hh + x4.x; { float yp = red16(hh * c4.x); if (n == 0) pys[i + 0] = yp; }
        hh = a4.y * hh + x4.y; { float yp = red16(hh * c4.y); if (n == 0) pys[i + 1] = yp; }
        hh = a4.z * hh + x4.z; { float yp = red16(hh * c4.z); if (n == 0) pys[i + 2] = yp; }
        hh = a4.w * hh + x4.w; { float yp = red16(hh * c4.w); if (n == 0) pys[i + 3] = yp; }
    }
}

// ---------------------------------------------------------------------------
extern "C" void kernel_launch(void* const* d_in, const int* in_sizes, int n_in,
                              void* d_out, int out_size, void* d_ws, size_t ws_size,
                              hipStream_t stream)
{
    (void)in_sizes; (void)n_in; (void)out_size; (void)ws_size;

    const float* x      = (const float*)d_in[0];
    const float* W_in   = (const float*)d_in[1];
    const float* conv_w = (const float*)d_in[2];
    const float* conv_b = (const float*)d_in[3];
    const float* W_x    = (const float*)d_in[4];
    const float* A_log  = (const float*)d_in[5];
    const float* Dvec   = (const float*)d_in[6];
    const float* W_out  = (const float*)d_in[7];
    float* out = (float*)d_out;

    // Workspace layout (~240 MB)
    __bf16* xz    = (__bf16*)d_ws;                              // [M][4096] bf16
    __bf16* xca   = xz + (size_t)M_TOTAL * (2 * D_INNER);       // [M][2048] bf16
    float*  dA_t  = (float*)(xca + (size_t)M_TOTAL * D_INNER);  // [64][SEQ]
    float*  dBx_t = dA_t + (size_t)BATCH * NSTATE * SEQ;
    float*  C_t   = dBx_t + (size_t)BATCH * NSTATE * SEQ;
    float*  ys    = C_t + (size_t)BATCH * NSTATE * SEQ;         // [M]
    __bf16* x_bf  = (__bf16*)(ys + M_TOTAL);                    // [M][1024]
    __bf16* Wi_bf = x_bf + (size_t)M_TOTAL * D_MODEL;           // [4096][1024]
    __bf16* Wo_bf = Wi_bf + (size_t)(2 * D_INNER) * D_MODEL;    // [1024][2048]

    // 0) one-shot f32 -> bf16 operand conversions
    cvt_bf16_kernel<<<(M_TOTAL * D_MODEL) / (256 * 8), 256, 0, stream>>>(x, x_bf);
    cvt_bf16_kernel<<<((2 * D_INNER) * D_MODEL) / (256 * 8), 256, 0, stream>>>(W_in, Wi_bf);
    cvt_bf16_kernel<<<(D_MODEL * D_INNER) / (256 * 8), 256, 0, stream>>>(W_out, Wo_bf);

    // 1) xz = x @ W_in^T  (M=16384, K=1024, N=4096)
    gemm_wmma<0><<<dim3((2 * D_INNER) / BN, M_TOTAL / BM), 256, 0, stream>>>(
        x_bf, Wi_bf, xz, nullptr, nullptr, nullptr, nullptr, nullptr,
        D_MODEL, 2 * D_INNER);

    // 2) causal depthwise conv + bias + silu -> xca
    conv_silu_kernel<<<(M_TOTAL * (D_INNER / 4)) / 256, 256, 0, stream>>>(
        xz, conv_w, conv_b, xca);

    // 3) x_dbl projection + scan prep (transposed layout)
    xdbl_kernel<<<M_TOTAL / 8, 256, 0, stream>>>(
        xca, W_x, A_log, dA_t, dBx_t, C_t);

    // 4) parallel linear recurrence -> ys
    scan_kernel<<<1, 1024, 0, stream>>>(dA_t, dBx_t, C_t, ys);

    // 5) out = [(ys + xc*D) * silu(z)] @ W_out^T  (M=16384, K=2048, N=1024)
    gemm_wmma<1><<<dim3(D_MODEL / BN, M_TOTAL / BM), 256, 0, stream>>>(
        nullptr, Wo_bf, nullptr, out, xca, xz, ys, Dvec,
        D_INNER, D_MODEL);
}
